// SchNetSimple_28544352649720
// MI455X (gfx1250) — matrix operand order repaired and checked by reading, not compiled, not run on hardware
//
#include <hip/hip_runtime.h>
#include <hip/hip_bf16.h>

#define HID 128
#define FILT 128
#define NGAUSS 51
#define NGPAD 64
#define NLAYER 6
#define NGRAPH 64

typedef __attribute__((ext_vector_type(16))) __bf16 bf16x16;
typedef __attribute__((ext_vector_type(8)))  float  f32x8;

union Frag {
  bf16x16 f;
  uint4 q[2];
  unsigned u[8];
};

__device__ __forceinline__ unsigned short f2bf(float x) {
  unsigned u = __builtin_bit_cast(unsigned, x);
  unsigned r = u + 0x7FFFu + ((u >> 16) & 1u);   // round-to-nearest-even
  return (unsigned short)(r >> 16);
}
__device__ __forceinline__ unsigned pack_bf2(float a, float b) {
  return (unsigned)f2bf(a) | ((unsigned)f2bf(b) << 16);
}

// ShiftedSoftplus: softplus(x) - log(2), numerically stable
__device__ __forceinline__ float sspf(float x) {
  return fmaxf(x, 0.0f) + __logf(1.0f + __expf(-fabsf(x))) - 0.69314718056f;
}

__device__ __forceinline__ f32x8 wmma_bf16(bf16x16 a, bf16x16 b, f32x8 c) {
  return __builtin_amdgcn_wmma_f32_16x16x32_bf16(false, a, false, b, (short)0, c,
                                                 false, false);
}

__device__ __forceinline__ void gatomic_add(float* p, float v) {
  __hip_atomic_fetch_add(p, v, __ATOMIC_RELAXED, __HIP_MEMORY_SCOPE_AGENT);
}

// A fragment (16x32 bf16, MxK) from row-major bf16; ldr in elements.
// ISA layout: lane l<16 holds row M=l, K=0..7 | K=16..23 (two contig 16B runs);
// lane l+16 holds same row at K=8..15 | K=24..31 -> two 128-bit loads per lane.
__device__ __forceinline__ bf16x16 load_afrag(const unsigned short* base, int ldr,
                                              int lane) {
  const int m = lane & 15, half = lane >> 4;
  const unsigned short* p = base + m * ldr + half * 8;
  Frag fa;
  fa.q[0] = *(const uint4*)(p);
  fa.q[1] = *(const uint4*)(p + 16);
  return fa.f;
}

// A fragment from row-major f32 with on-the-fly bf16 conversion (float2 loads).
__device__ __forceinline__ bf16x16 load_afrag_f32(const float* base, int ldr,
                                                  int lane) {
  const int m = lane & 15, half = lane >> 4;
  const float* p = base + m * ldr + half * 8;
  Frag fa;
#pragma unroll
  for (int v = 0; v < 8; ++v) {
    const int k = (v < 4) ? 2 * v : 16 + 2 * (v - 4);
    const float2 t = *(const float2*)(p + k);
    fa.u[v] = pack_bf2(t.x, t.y);
  }
  return fa.f;
}

// B fragment (32x16 bf16, KxN) from pre-swizzled LDS.
// Storage order: dword index = lane*8 + v  ->  each lane reads 32 contiguous
// bytes = two ds_read_b128.
__device__ __forceinline__ bf16x16 load_bfrag(const unsigned* frag, int lane) {
  const uint4* p = (const uint4*)(frag + lane * 8);
  Frag fb;
  fb.q[0] = p[0];
  fb.q[1] = p[1];
  return fb.f;
}

// Swizzle [Ksrc,N] row-major f32 weights into B-fragment order (bf16),
// zero-padding K up to Kpad. Fragment (kt,nt) at fragbuf[(kt*(N/16)+nt)*256].
// Within a fragment: dword d = lane*8+v holds W[k..k+1][n] with
// n = nt*16 + (lane&15), k = kt*32 + (lane>>4)*8 + v*2 + (v>=4 ? 8 : 0).
__device__ void swizzle_weights(const float* W, int Ksrc, int Kpad, int N,
                                unsigned* fragbuf, int tid, int nthreads) {
  const int NT = N / 16;
  const int total = (Kpad / 32) * NT * 256;
  for (int d = tid; d < total; d += nthreads) {
    const int v = d & 7;
    const int lane = (d >> 3) & 31;
    const int frag = d >> 8;
    const int nt = frag % NT, kt = frag / NT;
    const int n = nt * 16 + (lane & 15);
    const int k = kt * 32 + (lane >> 4) * 8 + v * 2 + ((v >= 4) ? 8 : 0);
    const float a = (k < Ksrc) ? W[k * N + n] : 0.0f;
    const float b = (k + 1 < Ksrc) ? W[(k + 1) * N + n] : 0.0f;
    fragbuf[d] = pack_bf2(a, b);
  }
}

// ---------------------------------------------------------------- small kernels

__global__ void k_embed(const int* __restrict__ z, const float* __restrict__ emb,
                        float* __restrict__ h, unsigned short* __restrict__ hbf,
                        int N) {
  const int i = blockIdx.x * blockDim.x + threadIdx.x;
  if (i >= N * HID) return;
  const int n = i >> 7;
  const float v = emb[(size_t)z[n] * HID + (i & 127)];
  h[i] = v;
  hbf[i] = f2bf(v);
}

__global__ void k_edge_pre(const float* __restrict__ pos, const int* __restrict__ ei,
                           float* __restrict__ Cc, unsigned short* __restrict__ attr,
                           int E) {
  const int e = blockIdx.x * blockDim.x + threadIdx.x;
  if (e >= E) return;
  const int r = ei[e], c = ei[E + e];
  const float dx = pos[r * 3 + 0] - pos[c * 3 + 0];
  const float dy = pos[r * 3 + 1] - pos[c * 3 + 1];
  const float dz = pos[r * 3 + 2] - pos[c * 3 + 2];
  const float d = sqrtf(dx * dx + dy * dy + dz * dz);
  Cc[e] = 0.5f * (__cosf(d * 0.31415926535f) + 1.0f);  // pi/cutoff, cutoff=10
  __attribute__((aligned(16))) unsigned buf[NGPAD / 2];
#pragma unroll
  for (int g = 0; g < NGPAD / 2; ++g) {
    const float d0 = d - (float)(2 * g) * 0.2f;        // offsets spaced 10/50
    const float d1 = d - (float)(2 * g + 1) * 0.2f;
    const float v0 = (2 * g < NGAUSS) ? __expf(-12.5f * d0 * d0) : 0.0f;
    const float v1 = (2 * g + 1 < NGAUSS) ? __expf(-12.5f * d1 * d1) : 0.0f;
    buf[g] = pack_bf2(v0, v1);
  }
  uint4* dst = (uint4*)(attr + (size_t)e * NGPAD);
  const uint4* src = (const uint4*)buf;
#pragma unroll
  for (int q = 0; q < NGPAD / 8; ++q) dst[q] = src[q];
}

__global__ void k_zero(float* __restrict__ p, int n) {
  const int i = blockIdx.x * blockDim.x + threadIdx.x;
  if (i < n) p[i] = 0.0f;
}

__global__ void k_counts(const int* __restrict__ batch, int* __restrict__ counts,
                         int N) {
  const int i = blockIdx.x * blockDim.x + threadIdx.x;
  if (i < N) atomicAdd(&counts[batch[i]], 1);
}

__global__ void k_final(const float* __restrict__ sums, const int* __restrict__ counts,
                        float* __restrict__ out) {
  const int i = blockIdx.x * blockDim.x + threadIdx.x;
  if (i >= NGRAPH * HID) return;
  out[i] = sums[i] / fmaxf((float)counts[i >> 7], 1.0f);
}

// ------------------------------------------------------------- node GEMM: xh=h@W
__global__ __launch_bounds__(256) void k_xh(const unsigned short* __restrict__ hbf,
                                            const float* __restrict__ w,
                                            float* __restrict__ xh, int nnodes) {
  __shared__ unsigned s_w[4 * 8 * 256];  // 32 KB fragment-swizzled weights
  const int tid = threadIdx.x;
  swizzle_weights(w, HID, HID, FILT, s_w, tid, blockDim.x);
  __syncthreads();
  const int lane = tid & 31, wave = tid >> 5;
  const int nlo = lane & 15, mhi = lane >> 4;
  const int ntiles = nnodes >> 4;
  for (int tile = blockIdx.x * 8 + wave; tile < ntiles; tile += gridDim.x * 8) {
    const int n0 = tile << 4;
    bf16x16 a[4];
#pragma unroll
    for (int kt = 0; kt < 4; ++kt)
      a[kt] = load_afrag(hbf + (size_t)n0 * HID + kt * 32, HID, lane);
    // single base pointer; all 64 stores use compile-time immediate offsets
    float* orow = xh + ((size_t)n0 + (size_t)mhi * 8) * FILT + nlo;
#pragma unroll
    for (int nt = 0; nt < 8; ++nt) {
      f32x8 acc = {};
#pragma unroll
      for (int kt = 0; kt < 4; ++kt)
        acc = wmma_bf16(a[kt], load_bfrag(s_w + (kt * 8 + nt) * 256, lane), acc);
#pragma unroll
      for (int v = 0; v < 8; ++v)
        orow[v * FILT + nt * 16] = acc[v];
    }
  }
}

// ---------------------------------------------------- fused per-edge filter kernel
// W = ssp(attr@W1+b1)@W2+b2, *cutoff; msg = xh[col]*W; agg[row] += msg (f32 atomics)
__global__ __launch_bounds__(256) void k_edge(
    const unsigned short* __restrict__ attr, const float* __restrict__ Cc,
    const int* __restrict__ rowp, const int* __restrict__ colp,
    const float* __restrict__ w1, const float* __restrict__ b1,
    const float* __restrict__ w2, const float* __restrict__ b2,
    const float* __restrict__ xh, float* __restrict__ agg, int nedges) {
  extern __shared__ char smem[];
  unsigned* s_w1 = (unsigned*)smem;                       // 16 KB (K=64 frags)
  unsigned* s_w2 = s_w1 + 2 * 8 * 256;                    // 32 KB (K=128 frags)
  float* s_b1 = (float*)(s_w2 + 4 * 8 * 256);
  float* s_b2 = s_b1 + FILT;
  unsigned short* s_t = (unsigned short*)(s_b2 + FILT);   // 8 waves * 16x128 bf16

  const int tid = threadIdx.x;
  swizzle_weights(w1, NGAUSS, NGPAD, FILT, s_w1, tid, blockDim.x);
  swizzle_weights(w2, FILT, FILT, FILT, s_w2, tid, blockDim.x);
  if (tid < FILT) { s_b1[tid] = b1[tid]; s_b2[tid] = b2[tid]; }
  __syncthreads();

  const int lane = tid & 31, wave = tid >> 5;
  const int nlo = lane & 15, mhi = lane >> 4;
  unsigned short* st = s_t + wave * (16 * FILT);
  unsigned short* strow = st + mhi * 8 * FILT + nlo;

  // tile-invariant biases: hoist out of the loop (one LDS read per nt)
  float b1v[8], b2v[8];
#pragma unroll
  for (int nt = 0; nt < 8; ++nt) {
    b1v[nt] = s_b1[nt * 16 + nlo];
    b2v[nt] = s_b2[nt * 16 + nlo];
  }

  const int ntiles = nedges >> 4;
  const int stride = gridDim.x * 8;
  for (int tile = blockIdx.x * 8 + wave; tile < ntiles; tile += stride) {
    const int e0 = tile << 4;
    const unsigned short* arow = attr + (size_t)e0 * NGPAD;
    __builtin_prefetch(arow + (size_t)stride * 16 * NGPAD, 0, 1);
    // ---- GEMM1: t = ssp(attr @ W1 + b1) -> restage bf16 tile in LDS ----
    const bf16x16 a0 = load_afrag(arow, NGPAD, lane);
    const bf16x16 a1 = load_afrag(arow + 32, NGPAD, lane);
#pragma unroll
    for (int nt = 0; nt < 8; ++nt) {
      f32x8 acc = {};
      acc = wmma_bf16(a0, load_bfrag(s_w1 + nt * 256, lane), acc);
      acc = wmma_bf16(a1, load_bfrag(s_w1 + (8 + nt) * 256, lane), acc);
#pragma unroll
      for (int v = 0; v < 8; ++v)
        strow[v * FILT + nt * 16] = f2bf(sspf(acc[v] + b1v[nt]));
    }
    // per-edge metadata + hoisted gather/scatter row pointers
    const int* rp = rowp + e0 + mhi * 8;
    const int* cp = colp + e0 + mhi * 8;
    const float* cc = Cc + e0 + mhi * 8;
    float Ce[8];
    const float* xptr[8];
    float* aptr[8];
#pragma unroll
    for (int v = 0; v < 8; ++v) {
      Ce[v] = cc[v];
      xptr[v] = xh + (size_t)cp[v] * FILT + nlo;
      aptr[v] = agg + (size_t)rp[v] * FILT + nlo;
    }
    // ---- GEMM2 + cutoff + gather/modulate/scatter ----
    bf16x16 a2[4];
#pragma unroll
    for (int kt = 0; kt < 4; ++kt)
      a2[kt] = load_afrag(st + kt * 32, FILT, lane);  // same-wave LDS: DS in-order
#pragma unroll
    for (int nt = 0; nt < 8; ++nt) {
      f32x8 acc = {};
#pragma unroll
      for (int kt = 0; kt < 4; ++kt)
        acc = wmma_bf16(a2[kt], load_bfrag(s_w2 + (kt * 8 + nt) * 256, lane), acc);
#pragma unroll
      for (int v = 0; v < 8; ++v) {
        const float Wv = (acc[v] + b2v[nt]) * Ce[v];
        const float xv = xptr[v][nt * 16];            // immediate offset gather
        gatomic_add(aptr[v] + nt * 16, xv * Wv);      // immediate offset scatter
      }
    }
  }
}

// ----------------------------------- node update: h += ssp(agg@cl2+b)@lin + b
__global__ __launch_bounds__(256) void k_node_update(
    const float* __restrict__ agg, const float* __restrict__ w2,
    const float* __restrict__ b2, const float* __restrict__ w3,
    const float* __restrict__ b3, float* __restrict__ h,
    unsigned short* __restrict__ hbf, int nnodes) {
  extern __shared__ char smem[];
  unsigned* s_w2 = (unsigned*)smem;
  unsigned* s_w3 = s_w2 + 4 * 8 * 256;
  float* s_b2 = (float*)(s_w3 + 4 * 8 * 256);
  float* s_b3 = s_b2 + HID;
  unsigned short* s_t = (unsigned short*)(s_b3 + HID);

  const int tid = threadIdx.x;
  swizzle_weights(w2, FILT, FILT, HID, s_w2, tid, blockDim.x);
  swizzle_weights(w3, HID, HID, HID, s_w3, tid, blockDim.x);
  if (tid < HID) { s_b2[tid] = b2[tid]; s_b3[tid] = b3[tid]; }
  __syncthreads();

  const int lane = tid & 31, wave = tid >> 5;
  const int nlo = lane & 15, mhi = lane >> 4;
  unsigned short* st = s_t + wave * (16 * HID);
  unsigned short* strow = st + mhi * 8 * HID + nlo;

  float b2v[8], b3v[8];
#pragma unroll
  for (int nt = 0; nt < 8; ++nt) {
    b2v[nt] = s_b2[nt * 16 + nlo];
    b3v[nt] = s_b3[nt * 16 + nlo];
  }

  const int ntiles = nnodes >> 4;
  for (int tile = blockIdx.x * 8 + wave; tile < ntiles; tile += gridDim.x * 8) {
    const int n0 = tile << 4;
    bf16x16 a[4];
#pragma unroll
    for (int kt = 0; kt < 4; ++kt)
      a[kt] = load_afrag_f32(agg + (size_t)n0 * FILT + kt * 32, FILT, lane);
#pragma unroll
    for (int nt = 0; nt < 8; ++nt) {
      f32x8 acc = {};
#pragma unroll
      for (int kt = 0; kt < 4; ++kt)
        acc = wmma_bf16(a[kt], load_bfrag(s_w2 + (kt * 8 + nt) * 256, lane), acc);
#pragma unroll
      for (int v = 0; v < 8; ++v)
        strow[v * HID + nt * 16] = f2bf(sspf(acc[v] + b2v[nt]));
    }
    bf16x16 a2[4];
#pragma unroll
    for (int kt = 0; kt < 4; ++kt)
      a2[kt] = load_afrag(st + kt * 32, HID, lane);
    float* hrow = h + ((size_t)n0 + (size_t)mhi * 8) * HID + nlo;
    unsigned short* hbrow = hbf + ((size_t)n0 + (size_t)mhi * 8) * HID + nlo;
#pragma unroll
    for (int nt = 0; nt < 8; ++nt) {
      f32x8 acc = {};
#pragma unroll
      for (int kt = 0; kt < 4; ++kt)
        acc = wmma_bf16(a2[kt], load_bfrag(s_w3 + (kt * 8 + nt) * 256, lane), acc);
#pragma unroll
      for (int v = 0; v < 8; ++v) {
        const int off = v * HID + nt * 16;
        const float nv = hrow[off] + acc[v] + b3v[nt];
        hrow[off] = nv;
        hbrow[off] = f2bf(nv);
      }
    }
  }
}

// ------------------- output MLP + pooled scatter: sums[batch] += ssp(h@W1+b)@W2+b
__global__ __launch_bounds__(256) void k_out(
    const unsigned short* __restrict__ hbf, const float* __restrict__ w1,
    const float* __restrict__ b1, const float* __restrict__ w2,
    const float* __restrict__ b2, const int* __restrict__ batch,
    float* __restrict__ sums, int nnodes) {
  extern __shared__ char smem[];
  unsigned* s_w1 = (unsigned*)smem;
  unsigned* s_w2 = s_w1 + 4 * 8 * 256;
  float* s_b1 = (float*)(s_w2 + 4 * 8 * 256);
  float* s_b2 = s_b1 + HID;
  unsigned short* s_t = (unsigned short*)(s_b2 + HID);

  const int tid = threadIdx.x;
  swizzle_weights(w1, HID, HID, HID, s_w1, tid, blockDim.x);
  swizzle_weights(w2, HID, HID, HID, s_w2, tid, blockDim.x);
  if (tid < HID) { s_b1[tid] = b1[tid]; s_b2[tid] = b2[tid]; }
  __syncthreads();

  const int lane = tid & 31, wave = tid >> 5;
  const int nlo = lane & 15, mhi = lane >> 4;
  unsigned short* st = s_t + wave * (16 * HID);
  unsigned short* strow = st + mhi * 8 * HID + nlo;

  float b1v[8], b2v[8];
#pragma unroll
  for (int nt = 0; nt < 8; ++nt) {
    b1v[nt] = s_b1[nt * 16 + nlo];
    b2v[nt] = s_b2[nt * 16 + nlo];
  }

  const int ntiles = nnodes >> 4;
  for (int tile = blockIdx.x * 8 + wave; tile < ntiles; tile += gridDim.x * 8) {
    const int n0 = tile << 4;
    bf16x16 a[4];
#pragma unroll
    for (int kt = 0; kt < 4; ++kt)
      a[kt] = load_afrag(hbf + (size_t)n0 * HID + kt * 32, HID, lane);
#pragma unroll
    for (int nt = 0; nt < 8; ++nt) {
      f32x8 acc = {};
#pragma unroll
      for (int kt = 0; kt < 4; ++kt)
        acc = wmma_bf16(a[kt], load_bfrag(s_w1 + (kt * 8 + nt) * 256, lane), acc);
#pragma unroll
      for (int v = 0; v < 8; ++v)
        strow[v * HID + nt * 16] = f2bf(sspf(acc[v] + b1v[nt]));
    }
    const int* bp = batch + n0 + mhi * 8;
    float* sptr[8];
#pragma unroll
    for (int v = 0; v < 8; ++v)
      sptr[v] = sums + (size_t)bp[v] * HID + nlo;
    bf16x16 a2[4];
#pragma unroll
    for (int kt = 0; kt < 4; ++kt)
      a2[kt] = load_afrag(st + kt * 32, HID, lane);
#pragma unroll
    for (int nt = 0; nt < 8; ++nt) {
      f32x8 acc = {};
#pragma unroll
      for (int kt = 0; kt < 4; ++kt)
        acc = wmma_bf16(a2[kt], load_bfrag(s_w2 + (kt * 8 + nt) * 256, lane), acc);
#pragma unroll
      for (int v = 0; v < 8; ++v)
        gatomic_add(sptr[v] + nt * 16, acc[v] + b2v[nt]);
    }
  }
}

// ---------------------------------------------------------------------- driver

extern "C" void kernel_launch(void* const* d_in, const int* in_sizes, int n_in,
                              void* d_out, int out_size, void* d_ws, size_t ws_size,
                              hipStream_t stream) {
  (void)n_in; (void)out_size; (void)ws_size;
  const int* z = (const int*)d_in[0];
  const float* pos = (const float*)d_in[1];
  const int* batch = (const int*)d_in[2];
  const int* ei = (const int*)d_in[3];
  const float* emb = (const float*)d_in[4];
  const float* mlp1_w = (const float*)d_in[5];
  const float* mlp1_b = (const float*)d_in[6];
  const float* mlp2_w = (const float*)d_in[7];
  const float* mlp2_b = (const float*)d_in[8];
  const float* cl1_w = (const float*)d_in[9];
  const float* cl2_w = (const float*)d_in[10];
  const float* cl2_b = (const float*)d_in[11];
  const float* lin_w = (const float*)d_in[12];
  const float* lin_b = (const float*)d_in[13];
  const float* out1_w = (const float*)d_in[14];
  const float* out1_b = (const float*)d_in[15];
  const float* out2_w = (const float*)d_in[16];
  const float* out2_b = (const float*)d_in[17];
  const int N = in_sizes[0];
  const int E = in_sizes[3] / 2;

  char* ws = (char*)d_ws;
  unsigned short* attr = (unsigned short*)ws;  ws += (size_t)E * NGPAD * 2;  // 64 MB
  float* Cc = (float*)ws;                      ws += (size_t)E * 4;          // 2 MB
  float* h = (float*)ws;                       ws += (size_t)N * HID * 4;    // 8 MB
  unsigned short* hbf = (unsigned short*)ws;   ws += (size_t)N * HID * 2;    // 4 MB
  float* xh = (float*)ws;                      ws += (size_t)N * FILT * 4;   // 8 MB
  float* agg = (float*)ws;                     ws += (size_t)N * FILT * 4;   // 8 MB
  float* sums = (float*)ws;                    ws += (size_t)NGRAPH * HID * 4;
  int* counts = (int*)ws;

  k_embed<<<(N * HID + 255) / 256, 256, 0, stream>>>(z, emb, h, hbf, N);
  k_edge_pre<<<(E + 255) / 256, 256, 0, stream>>>(pos, ei, Cc, attr, E);

  const size_t smem_edge = (2 * 8 * 256 + 4 * 8 * 256) * 4 + 2 * FILT * 4 +
                           8 * 16 * FILT * 2;  // 82944 B
  const size_t smem_node = (4 * 8 * 256 + 4 * 8 * 256) * 4 + 2 * HID * 4 +
                           8 * 16 * HID * 2;   // 99328 B

  for (int i = 0; i < NLAYER; ++i) {
    k_zero<<<(N * FILT + 255) / 256, 256, 0, stream>>>(agg, N * FILT);
    k_xh<<<64, 256, 0, stream>>>(hbf, cl1_w + (size_t)i * HID * FILT, xh, N);
    k_edge<<<1024, 256, smem_edge, stream>>>(
        attr, Cc, ei, ei + E, mlp1_w + (size_t)i * NGAUSS * FILT,
        mlp1_b + (size_t)i * FILT, mlp2_w + (size_t)i * FILT * FILT,
        mlp2_b + (size_t)i * FILT, xh, agg, E);
    k_node_update<<<64, 256, smem_node, stream>>>(
        agg, cl2_w + (size_t)i * FILT * HID, cl2_b + (size_t)i * HID,
        lin_w + (size_t)i * HID * HID, lin_b + (size_t)i * HID, h, hbf, N);
  }
  // zero sums (8192 floats) and the adjacent counts (64 ints; 0 bits == 0.0f)
  k_zero<<<(NGRAPH * HID + NGRAPH + 255) / 256, 256, 0, stream>>>(
      sums, NGRAPH * HID + NGRAPH);
  k_counts<<<(N + 255) / 256, 256, 0, stream>>>(batch, counts, N);
  k_out<<<64, 256, smem_node, stream>>>(hbf, out1_w, out1_b, out2_w, out2_b, batch,
                                        sums, N);
  k_final<<<(NGRAPH * HID + 255) / 256, 256, 0, stream>>>(sums, counts,
                                                          (float*)d_out);
}
